// MultiHeadSelfAtt_34024730919257
// MI455X (gfx1250) — matrix operand order repaired and checked by reading, not compiled, not run on hardware
//
#include <hip/hip_runtime.h>

// Problem constants (reference: B=4, S=2048, D=1024, H=16, DK=64)
#define BB  4
#define SS  2048
#define DDm 1024
#define HHn 16
#define DKK 64

typedef __attribute__((ext_vector_type(16))) __bf16 v16bf;
typedef __attribute__((ext_vector_type(2)))  __bf16 v2bf;
typedef __attribute__((ext_vector_type(8)))  float  v8f;
typedef __attribute__((ext_vector_type(4)))  unsigned int u32x4;
typedef __attribute__((ext_vector_type(8)))  int i32x8;
typedef __attribute__((ext_vector_type(4)))  int i32x4;

#if defined(__gfx1250__) && __has_builtin(__builtin_amdgcn_tensor_load_to_lds)
#define HAVE_TDM 1
#else
#define HAVE_TDM 0
#endif

union BfPack { unsigned int u32; unsigned short u16[2]; __bf16 bf[2]; v2bf v2; };

// f32 -> bf16, round-half-up ((u+0x8000)>>16): 2 VALU ops; bias vs RNE is ties-only,
// far below bf16's 2^-8 quantization noise.
__device__ __forceinline__ unsigned short f2bf(float f) {
  return (unsigned short)((__float_as_uint(f) + 0x8000u) >> 16);
}
__device__ __forceinline__ float bf2f(unsigned short h) {
  return __uint_as_float(((unsigned int)h) << 16);
}
// Packed f32x2 -> bf16x2 (hardware cvt when available; else cheap shift/mask pack)
__device__ __forceinline__ unsigned int f2bf_pk(float a, float b) {
#if __has_builtin(__builtin_amdgcn_cvt_pk_bf16_f32)
  BfPack p; p.v2 = __builtin_amdgcn_cvt_pk_bf16_f32(a, b);
  return p.u32;
#else
  unsigned int ra = __float_as_uint(a) + 0x8000u;
  unsigned int rb = __float_as_uint(b) + 0x8000u;
  return (ra >> 16) | (rb & 0xFFFF0000u);
#endif
}

// ---- WMMA fragment assembly per CDNA5 wave32 layouts (CDNA5 ISA) ----
// A (16x32 bf16): lane L: m = L&15, half = L>>4; VGPR j: K = 2*(j&3) + 16*(j>>2) + 8*half
__device__ __forceinline__ v16bf load_afrag(const unsigned short* tile, int ld) {
  const int lane = threadIdx.x & 31;
  const int half = lane >> 4, m = lane & 15;
  v16bf a;
#pragma unroll
  for (int j = 0; j < 8; ++j) {
    int k = ((j & 3) << 1) + ((j >> 2) << 4) + (half << 3);
    BfPack p; p.u32 = *(const unsigned int*)(tile + m * ld + k);
    a[2 * j]     = p.bf[0];
    a[2 * j + 1] = p.bf[1];
  }
  return a;
}
// B (32x16 bf16) from an [n][k] LDS tile: lane L: n = L&15, VGPR c: K = 2*(c&3) + 16*(c>>2) + 8*half
__device__ __forceinline__ v16bf load_bfrag_nmajor(const unsigned short* tile, int ld) {
  const int lane = threadIdx.x & 31;
  const int half = lane >> 4, n = lane & 15;
  const unsigned short* base = tile + n * ld + (half << 3);
  v16bf b;
#pragma unroll
  for (int c = 0; c < 8; ++c) {
    BfPack p; p.u32 = *(const unsigned int*)(base + ((c & 3) << 1) + ((c >> 2) << 4));
    b[2 * c]     = p.bf[0];
    b[2 * c + 1] = p.bf[1];
  }
  return b;
}

__device__ __forceinline__ v8f wmma_bf16(v16bf a, v16bf b, v8f c) {
  return __builtin_amdgcn_wmma_f32_16x16x32_bf16(false, a, false, b, (short)0, c,
                                                 false, false);
}

// Row reductions across the 16 lanes of a half-wave (C-tile rows striped over n-lanes)
__device__ __forceinline__ float rowmax16(float x) {
#pragma unroll
  for (int o = 1; o < 16; o <<= 1) x = fmaxf(x, __shfl_xor(x, o, 16));
  return x;
}
__device__ __forceinline__ float rowsum16(float x) {
#pragma unroll
  for (int o = 1; o < 16; o <<= 1) x += __shfl_xor(x, o, 16);
  return x;
}

#if HAVE_TDM
// 1-D TDM DMA: copy `nelem` bf16 (nelem*2 bytes) from global into LDS
// (descriptor packing per CDNA5 ISA; nelem < 65536)
__device__ __forceinline__ void tdm_load_lin(unsigned lds_off, const void* gptr,
                                             unsigned nelem) {
  unsigned long long ga = (unsigned long long)(size_t)gptr;
  u32x4 g0;
  g0[0] = 1u;                                               // count=1 (valid), user mode
  g0[1] = lds_off;                                          // LDS byte address
  g0[2] = (unsigned)ga;                                     // global addr lo
  g0[3] = (unsigned)((ga >> 32) & 0x1FFFFFFu) | (2u << 30); // addr hi | type=2 (image)
  i32x8 g1;
  g1[0] = (int)(1u << 16);        // workgroup_mask=0 (not in cluster), data_size=1 (2B)
  g1[1] = (int)(nelem << 16);     // tensor_dim0 low 16 (bits [63:48])
  g1[2] = (int)((nelem >> 16) | (1u << 16));  // tensor_dim0 hi | tensor_dim1 = 1
  g1[3] = (int)(nelem << 16);     // tensor_dim1 hi = 0 | tile_dim0 = nelem
  g1[4] = 0;                      // tile_dim1 = 0 (1-D), tile_dim2 = 0
  g1[5] = (int)nelem;             // tensor_dim0_stride lo
  g1[6] = 0;
  g1[7] = 0;
  i32x4 z4 = {0, 0, 0, 0};
#if __clang_major__ >= 23
  i32x8 z8 = {0, 0, 0, 0, 0, 0, 0, 0};
  __builtin_amdgcn_tensor_load_to_lds(g0, g1, z4, z4, z8, 0);
#else
  __builtin_amdgcn_tensor_load_to_lds(g0, g1, z4, z4, 0);
#endif
}
__device__ __forceinline__ void wait_tensor0() {
#if __has_builtin(__builtin_amdgcn_s_wait_tensorcnt)
  __builtin_amdgcn_s_wait_tensorcnt(0);
#else
  asm volatile("s_wait_tensorcnt 0x0" ::: "memory");
#endif
}
__device__ __forceinline__ void wait_tensor1() {
#if __has_builtin(__builtin_amdgcn_s_wait_tensorcnt)
  __builtin_amdgcn_s_wait_tensorcnt(1);
#else
  asm volatile("s_wait_tensorcnt 0x1" ::: "memory");
#endif
}
#endif  // HAVE_TDM

// ======================= Projection: out = (X @ W^T + b) * scale =======================
// Grid: (M/64, N/64), block 128 (4 waves). Output bf16 in head-split [B,H,S,DK] layout.
__global__ __launch_bounds__(128)
void mhsa_proj_kernel(const float* __restrict__ X, const float* __restrict__ W,
                      const float* __restrict__ bias, unsigned short* __restrict__ out,
                      float scale) {
  __shared__ unsigned short Xs[64 * 32];
  __shared__ unsigned short Ws[64 * 32];
  const int tid  = threadIdx.x;
  const int w    = tid >> 5;
  const int lane = tid & 31;
  const int half = lane >> 4, nl = lane & 15;
  const int m0 = blockIdx.x * 64;
  const int n0 = blockIdx.y * 64;

  v8f zero = {};
  v8f acc[4] = {zero, zero, zero, zero};

  for (int k0 = 0; k0 < DDm; k0 += 32) {
    __syncthreads();
#pragma unroll
    for (int t = tid; t < 64 * 16; t += 128) {  // f32 pairs -> packed bf16 u32
      int r = t >> 4, c2 = (t & 15) << 1;
      const float* xp = X + (size_t)(m0 + r) * DDm + k0 + c2;
      const float* wp = W + (size_t)(n0 + r) * DDm + k0 + c2;
      ((unsigned int*)Xs)[t] = f2bf_pk(xp[0], xp[1]);
      ((unsigned int*)Ws)[t] = f2bf_pk(wp[0], wp[1]);
    }
    __syncthreads();
    v16bf a = load_afrag(Xs + w * 16 * 32, 32);
    v16bf b0 = load_bfrag_nmajor(Ws + 0 * 16 * 32, 32);
    v16bf b1 = load_bfrag_nmajor(Ws + 1 * 16 * 32, 32);
    v16bf b2 = load_bfrag_nmajor(Ws + 2 * 16 * 32, 32);
    v16bf b3 = load_bfrag_nmajor(Ws + 3 * 16 * 32, 32);
    acc[0] = wmma_bf16(a, b0, acc[0]);
    acc[1] = wmma_bf16(a, b1, acc[1]);
    acc[2] = wmma_bf16(a, b2, acc[2]);
    acc[3] = wmma_bf16(a, b3, acc[3]);
  }

#pragma unroll
  for (int nt = 0; nt < 4; ++nt) {
    int col = n0 + nt * 16 + nl;
    float bv = bias[col];
    int h = col >> 6, dk = col & 63;
#pragma unroll
    for (int r = 0; r < 8; ++r) {
      int m = m0 + w * 16 + r + half * 8;
      int b = m >> 11;  // S = 2048
      int s = m & (SS - 1);
      float v = (acc[nt][r] + bv) * scale;
      out[(((size_t)b * HHn + h) * SS + s) * DKK + dk] = f2bf(v);
    }
  }
}

// ======================= V column sums (all rows / masked rows) =======================
__global__ __launch_bounds__(64)
void mhsa_vstat_kernel(const unsigned short* __restrict__ vb,
                       const unsigned char* __restrict__ mask,
                       float* __restrict__ vsum, float* __restrict__ vmsum) {
  int bh = blockIdx.x;
  int b  = bh >> 4;  // H = 16
  int dk = threadIdx.x;
  const unsigned short* vp = vb + (size_t)bh * SS * DKK + dk;
  const unsigned char*  mp = mask + (size_t)b * SS;
  float s = 0.f, sm = 0.f;
  for (int t = 0; t < SS; ++t) {
    float v = bf2f(vp[(size_t)t * DKK]);
    s += v;
    if (mp[t]) sm += v;
  }
  vsum [(size_t)bh * DKK + dk] = s;
  vmsum[(size_t)bh * DKK + dk] = sm;
}

// ======================= Attention (flash-style streaming softmax) =======================
// Grid: (S/64, B*H), block 128 (4 waves, each wave owns 16 query rows).
// Q and K tiles arrive via TDM (K double-buffered, one tile ahead); V transposed into LDS.
__global__ __launch_bounds__(128)
void mhsa_attn_kernel(const unsigned short* __restrict__ qb,
                      const unsigned short* __restrict__ kb,
                      const unsigned short* __restrict__ vb,
                      const unsigned char* __restrict__ mask,
                      const float* __restrict__ vsum,
                      const float* __restrict__ vmsum,
                      float* __restrict__ out) {
  __shared__ unsigned short Qs[64 * 64];
  __shared__ unsigned short Ks[2][32 * 64];   // [seq][dk], double-buffered for TDM
  __shared__ unsigned short Vt[64 * 32];      // transposed: [dk][seq]
  __shared__ unsigned short Ps[4][16 * 32];
  __shared__ float mkeep[32];

  const int tid  = threadIdx.x;
  const int w    = tid >> 5;
  const int lane = tid & 31;
  const int half = lane >> 4, nl = lane & 15;
  const int m0 = blockIdx.x * 64;
  const int bh = blockIdx.y;
  const int b  = bh >> 4;
  const int h  = bh & 15;

  const size_t base = (size_t)bh * SS * DKK;
  const unsigned short* kbase = kb + base;

#if HAVE_TDM
  if (w == 0) {
    tdm_load_lin((unsigned)(size_t)(const void*)&Qs[0],
                 qb + base + (size_t)m0 * DKK, 64 * 64);      // 8KB Q tile
    tdm_load_lin((unsigned)(size_t)(const void*)&Ks[0][0], kbase, 32 * 64);
    wait_tensor1();  // in-order: Q done; K tile 0 may still be in flight
  }
#else
  {  // manual Q staging
    const unsigned int* src = (const unsigned int*)(qb + base + (size_t)m0 * DKK);
    unsigned int* dst = (unsigned int*)Qs;
    for (int t = tid; t < 64 * 32; t += 128) dst[t] = src[t];
  }
#endif
  __syncthreads();
  v16bf aq0 = load_afrag(Qs + w * 16 * 64, 64);       // dk 0..31
  v16bf aq1 = load_afrag(Qs + w * 16 * 64 + 32, 64);  // dk 32..63

  v8f zero = {};
  v8f acc[4] = {zero, zero, zero, zero};
  float mrow[8], lrow[8];
#pragma unroll
  for (int r = 0; r < 8; ++r) { mrow[r] = -1e30f; lrow[r] = 0.f; }

  for (int j = 0; j < SS / 32; ++j) {
    __syncthreads();  // all waves done reading Vt and Ks[(j+1)&1] (from iter j-1)
    {  // stage V tile j transposed: Vt[dk][seq]  (global uint read -> 2 u16 stores)
      const unsigned int* vsrc = (const unsigned int*)(vb + base + (size_t)j * 32 * DKK);
      for (int t = tid; t < 32 * 32; t += 128) {
        int seq = t >> 5, dkp = t & 31;
        unsigned u = vsrc[t];
        Vt[(2 * dkp)     * 32 + seq] = (unsigned short)u;
        Vt[(2 * dkp + 1) * 32 + seq] = (unsigned short)(u >> 16);
      }
      if (tid < 32) mkeep[tid] = mask[(size_t)b * SS + j * 32 + tid] ? 0.f : 1.f;
    }
#if HAVE_TDM
    if (w == 0) {
      if (j + 1 < SS / 32) {  // pipeline: issue tile j+1, wait for tile j
        tdm_load_lin((unsigned)(size_t)(const void*)&Ks[(j + 1) & 1][0],
                     kbase + (size_t)(j + 1) * 32 * DKK, 32 * 64);
        wait_tensor1();
      } else {
        wait_tensor0();
      }
    }
    const unsigned short* Kcur = Ks[j & 1];
#else
    {  // fallback: manual K staging
      const unsigned int* ksrc = (const unsigned int*)(kbase + (size_t)j * 32 * DKK);
      unsigned int* kdst = (unsigned int*)Ks[0];
      for (int t = tid; t < 32 * 32; t += 128) kdst[t] = ksrc[t];
    }
    const unsigned short* Kcur = Ks[0];
#endif
    __syncthreads();
    if (j + 1 < SS / 32)
      __builtin_prefetch(vb + base + (size_t)(j + 1) * 32 * DKK, 0, 1);

    // scores for the two 16-column subtiles (1/D scale already folded into Q)
    v16bf bk00 = load_bfrag_nmajor(Kcur, 64);
    v16bf bk01 = load_bfrag_nmajor(Kcur + 32, 64);
    v16bf bk10 = load_bfrag_nmajor(Kcur + 16 * 64, 64);
    v16bf bk11 = load_bfrag_nmajor(Kcur + 16 * 64 + 32, 64);
    v8f c0 = zero, c1 = zero;
    c0 = wmma_bf16(aq0, bk00, c0);
    c1 = wmma_bf16(aq0, bk10, c1);
    c0 = wmma_bf16(aq1, bk01, c0);
    c1 = wmma_bf16(aq1, bk11, c1);

    float keep0 = mkeep[nl], keep1 = mkeep[16 + nl];
    unsigned short* pw = Ps[w];
#pragma unroll
    for (int r = 0; r < 8; ++r) {
      float s0 = c0[r], s1 = c1[r];
      float rm   = rowmax16(fmaxf(s0, s1));
      float mnew = fmaxf(mrow[r], rm);
      float f    = __expf(mrow[r] - mnew);
      float p0   = __expf(s0 - mnew);
      float p1   = __expf(s1 - mnew);
      lrow[r] = lrow[r] * f + rowsum16(p0 + p1);  // denominator over ALL columns
      mrow[r] = mnew;
#pragma unroll
      for (int nt = 0; nt < 4; ++nt) acc[nt][r] *= f;
      int rl = r + half * 8;
      // masked columns contribute 0 to P@V (their 1e-15 term handled via vmsum)
      pw[rl * 32 + nl]      = f2bf(p0 * keep0);
      pw[rl * 32 + 16 + nl] = f2bf(p1 * keep1);
    }

    asm volatile("s_wait_dscnt 0" ::: "memory");  // P stores -> A-fragment loads (same wave)
    v16bf ap  = load_afrag(pw, 32);
    v16bf bv0 = load_bfrag_nmajor(Vt + 0 * 16 * 32, 32);  // B[k=seq][n=dk] via transpose
    v16bf bv1 = load_bfrag_nmajor(Vt + 1 * 16 * 32, 32);
    v16bf bv2 = load_bfrag_nmajor(Vt + 2 * 16 * 32, 32);
    v16bf bv3 = load_bfrag_nmajor(Vt + 3 * 16 * 32, 32);
    acc[0] = wmma_bf16(ap, bv0, acc[0]);
    acc[1] = wmma_bf16(ap, bv1, acc[1]);
    acc[2] = wmma_bf16(ap, bv2, acc[2]);
    acc[3] = wmma_bf16(ap, bv3, acc[3]);
  }

  // epilogue: out = P/l @ V + 1e-15 * (masked V sum); fully-masked rows = 1e-15 * sum(V)
#pragma unroll
  for (int r = 0; r < 8; ++r) {
    int s = m0 + w * 16 + r + half * 8;
    bool mq = mask[(size_t)b * SS + s] != 0;
    float inv_l = 1.0f / lrow[r];
#pragma unroll
    for (int nt = 0; nt < 4; ++nt) {
      int dk = nt * 16 + nl;
      float vs  = vsum [(size_t)bh * DKK + dk];
      float vms = vmsum[(size_t)bh * DKK + dk];
      float o = mq ? (1e-15f * vs) : (acc[nt][r] * inv_l + 1e-15f * vms);
      out[((size_t)b * SS + s) * DDm + h * DKK + dk] = o;
    }
  }
}

extern "C" void kernel_launch(void* const* d_in, const int* in_sizes, int n_in,
                              void* d_out, int out_size, void* d_ws, size_t ws_size,
                              hipStream_t stream) {
  const float* key   = (const float*)d_in[0];
  const float* query = (const float*)d_in[1];
  const float* value = (const float*)d_in[2];
  const unsigned char* mask = (const unsigned char*)d_in[3];  // jnp.bool_ = 1 byte/elem
  const float* Wq = (const float*)d_in[4];
  const float* bq = (const float*)d_in[5];
  const float* Wk = (const float*)d_in[6];
  const float* bk = (const float*)d_in[7];
  const float* Wv = (const float*)d_in[8];
  const float* bv = (const float*)d_in[9];

  const size_t MN = (size_t)BB * SS * DDm;  // elements per projection
  unsigned short* qb = (unsigned short*)d_ws;
  unsigned short* kb = qb + MN;
  unsigned short* vb = kb + MN;
  float* vsum  = (float*)(vb + MN);
  float* vmsum = vsum + (size_t)BB * HHn * DKK;

  dim3 gridP(BB * SS / 64, DDm / 64);  // (128, 16)
  // Q projection: fold non-standard 1/D scale (exactly 2^-10, lossless in bf16)
  mhsa_proj_kernel<<<gridP, 128, 0, stream>>>(query, Wq, bq, qb, 1.0f / (float)DDm);
  mhsa_proj_kernel<<<gridP, 128, 0, stream>>>(key,   Wk, bk, kb, 1.0f);
  mhsa_proj_kernel<<<gridP, 128, 0, stream>>>(value, Wv, bv, vb, 1.0f);

  mhsa_vstat_kernel<<<BB * HHn, 64, 0, stream>>>(vb, mask, vsum, vmsum);

  dim3 gridA(SS / 64, BB * HHn);  // (32, 64)
  mhsa_attn_kernel<<<gridA, 128, 0, stream>>>(qb, kb, vb, mask, vsum, vmsum,
                                              (float*)d_out);
}
